// CausalSelfAttention_57569741635981
// MI455X (gfx1250) — matrix operand order repaired and checked by reading, not compiled, not run on hardware
//
#include <hip/hip_runtime.h>
#include <math.h>

typedef _Float16 half_t;
typedef _Float16 v16h __attribute__((ext_vector_type(16)));
typedef _Float16 v8h  __attribute__((ext_vector_type(8)));
typedef float    v8f  __attribute__((ext_vector_type(8)));

#define BB 2
#define TT 2048
#define CC 768
#define HH 12
#define DD 64
#define MM (BB*TT)          /* 4096 */
#define NQKV (3*CC)         /* 2304 */

/* workspace layout, offsets in half_t elements */
#define OFF_XH   0
#define OFF_WAH  (OFF_XH  + MM*CC)
#define OFF_WPH  (OFF_WAH + NQKV*CC)
#define OFF_QH   (OFF_WPH + CC*CC)
#define OFF_KH   (OFF_QH  + MM*CC)
#define OFF_VTH  (OFF_KH  + MM*CC)
#define OFF_CTX  (OFF_VTH + MM*CC)

static __device__ __forceinline__ v16h ld2g(const half_t* p, int delta) {
  union { struct { v8h lo; v8h hi; } s; v16h v; } u;
  u.s.lo = *(const v8h*)(p);
  u.s.hi = *(const v8h*)(p + delta);
  return u.v;
}

static __device__ __forceinline__ v8f wmma_f16(v16h a, v16h b, v8f c) {
  return __builtin_amdgcn_wmma_f32_16x16x32_f16(false, a, false, b, (short)0, c,
                                                false, false);
}

/* ---------------- fp32 -> fp16 conversion of all operands ---------------- */
__global__ void __launch_bounds__(256) k_convert(const float* __restrict__ x,
                                                 const float* __restrict__ wa,
                                                 const float* __restrict__ wp,
                                                 half_t* __restrict__ ws) {
  const int n0 = MM * CC, n1 = NQKV * CC, n2 = CC * CC;
  const int total = n0 + n1 + n2;
  for (int i = blockIdx.x * blockDim.x + threadIdx.x; i < total;
       i += gridDim.x * blockDim.x) {
    if (i < n0)            ws[OFF_XH  + i]             = (half_t)x[i];
    else if (i < n0 + n1)  ws[OFF_WAH + (i - n0)]      = (half_t)wa[i - n0];
    else                   ws[OFF_WPH + (i - n0 - n1)] = (half_t)wp[i - n0 - n1];
  }
}

/* ---- 32x64 output tile of Out = A[M,K] * W[N,K]^T, software pipelined ---- */
template <int KDIM>
static __device__ __forceinline__ void gemm32x64(const half_t* __restrict__ A0,
                                                 const half_t* __restrict__ B0,
                                                 v8f acc[8]) {
  const int lane = threadIdx.x & 31;
  const int l15  = lane & 15;
  const int hi   = lane >> 4;
  const half_t* ap0 = A0 + l15 * KDIM + hi * 16;         /* rows 0..15  */
  const half_t* ap1 = A0 + (16 + l15) * KDIM + hi * 16;  /* rows 16..31 */
  const half_t* bp  = B0 + l15 * KDIM + hi * 16;

  /* prologue: load K-step 0 */
  v16h a0 = ld2g(ap0, 8);
  v16h a1 = ld2g(ap1, 8);
  v16h b0 = ld2g(bp + 0 * 16 * KDIM, 8);
  v16h b1 = ld2g(bp + 1 * 16 * KDIM, 8);
  v16h b2 = ld2g(bp + 2 * 16 * KDIM, 8);
  v16h b3 = ld2g(bp + 3 * 16 * KDIM, 8);

#pragma unroll 2
  for (int kk = 32; kk < KDIM; kk += 32) {
    /* issue next K-step loads first: stay in flight under the WMMAs */
    v16h na0 = ld2g(ap0 + kk, 8);
    v16h na1 = ld2g(ap1 + kk, 8);
    v16h nb0 = ld2g(bp + 0 * 16 * KDIM + kk, 8);
    v16h nb1 = ld2g(bp + 1 * 16 * KDIM + kk, 8);
    v16h nb2 = ld2g(bp + 2 * 16 * KDIM + kk, 8);
    v16h nb3 = ld2g(bp + 3 * 16 * KDIM + kk, 8);

    acc[0] = wmma_f16(a0, b0, acc[0]);
    acc[1] = wmma_f16(a0, b1, acc[1]);
    acc[2] = wmma_f16(a0, b2, acc[2]);
    acc[3] = wmma_f16(a0, b3, acc[3]);
    acc[4] = wmma_f16(a1, b0, acc[4]);
    acc[5] = wmma_f16(a1, b1, acc[5]);
    acc[6] = wmma_f16(a1, b2, acc[6]);
    acc[7] = wmma_f16(a1, b3, acc[7]);

    a0 = na0; a1 = na1; b0 = nb0; b1 = nb1; b2 = nb2; b3 = nb3;
  }
  /* epilogue: last K-step */
  acc[0] = wmma_f16(a0, b0, acc[0]);
  acc[1] = wmma_f16(a0, b1, acc[1]);
  acc[2] = wmma_f16(a0, b2, acc[2]);
  acc[3] = wmma_f16(a0, b3, acc[3]);
  acc[4] = wmma_f16(a1, b0, acc[4]);
  acc[5] = wmma_f16(a1, b1, acc[5]);
  acc[6] = wmma_f16(a1, b2, acc[6]);
  acc[7] = wmma_f16(a1, b3, acc[7]);
}

/* ------------------------------ QKV GEMM ------------------------------- */
__global__ void __launch_bounds__(256) k_gemm_qkv(half_t* __restrict__ ws) {
  const half_t* Xh  = ws + OFF_XH;
  const half_t* Wah = ws + OFF_WAH;
  half_t* Qh  = ws + OFF_QH;
  half_t* Kh  = ws + OFF_KH;
  half_t* Vth = ws + OFF_VTH;

  const int wid = blockIdx.x * (blockDim.x >> 5) + (threadIdx.x >> 5);
  const int tm = wid & 127;        /* 128 m-tiles (32 rows each) */
  const int tn = wid >> 7;         /* 36  n-tiles (64 cols each) */
  const int m0 = tm * 32, n0 = tn * 64;

  v8f acc[8];
#pragma unroll
  for (int i = 0; i < 8; ++i)
#pragma unroll
    for (int j = 0; j < 8; ++j) acc[i][j] = 0.0f;

  gemm32x64<CC>(Xh + m0 * CC, Wah + n0 * CC, acc);

  const int lane = threadIdx.x & 31, l15 = lane & 15, hi = lane >> 4;
#pragma unroll
  for (int mi = 0; mi < 2; ++mi) {
#pragma unroll
    for (int g = 0; g < 4; ++g) {
      const int o   = n0 + g * 16 + l15;
      const int sec = o >> 6;        /* head section 0..35 */
      const int d   = o & 63;
#pragma unroll
      for (int r = 0; r < 8; ++r) {
        const int m = m0 + mi * 16 + r + hi * 8;
        const int t = m & (TT - 1);
        const int b = m >> 11;
        const half_t hv = (half_t)acc[mi * 4 + g][r];
        if (sec < 12) {
          const int bh = b * HH + sec;
          Qh[(bh * TT + t) * DD + d] = hv;
        } else if (sec < 24) {
          const int bh = b * HH + (sec - 12);
          Kh[(bh * TT + t) * DD + d] = hv;
        } else {
          const int bh = b * HH + (sec - 24);
          Vth[(bh * DD + d) * TT + t] = hv;   /* V stored transposed */
        }
      }
    }
  }
}

/* --------------------- flash attention, 1 wave/q-tile -------------------- */
__global__ void __launch_bounds__(256) k_attention(half_t* __restrict__ ws) {
  __shared__ __align__(16) half_t pbuf[8][16 * 32];

  const half_t* Qh  = ws + OFF_QH;
  const half_t* Kh  = ws + OFF_KH;
  const half_t* Vth = ws + OFF_VTH;
  half_t* Ctx = ws + OFF_CTX;

  const int wslot = threadIdx.x >> 5;
  const int wid   = blockIdx.x * 8 + wslot;   /* 0..3071 */
  const int qt    = wid & 127;
  const int bh    = wid >> 7;                 /* 0..23 */
  const int q0    = qt * 16;

  const int lane = threadIdx.x & 31, l15 = lane & 15, hi = lane >> 4;
  const half_t* Qb = Qh  + bh * (TT * DD);
  const half_t* Kb = Kh  + bh * (TT * DD);
  const half_t* Vb = Vth + bh * (TT * DD);
  half_t* myp = pbuf[wslot];
  const float NEG_INF = -__builtin_inff();

  /* Q A-tiles (16 rows x 64 head dims = two K-steps) held in registers */
  v16h qa[2];
#pragma unroll
  for (int s = 0; s < 2; ++s)
    qa[s] = ld2g(Qb + (q0 + l15) * DD + s * 32 + hi * 16, 8);

  v8f oacc[4];
#pragma unroll
  for (int g = 0; g < 4; ++g)
#pragma unroll
    for (int j = 0; j < 8; ++j) oacc[g][j] = 0.0f;
  float m8[8], l8[8];
#pragma unroll
  for (int r = 0; r < 8; ++r) { m8[r] = NEG_INF; l8[r] = 0.0f; }

  const int nkb = (q0 + 15) / 32 + 1;
  for (int kb = 0; kb < nkb; ++kb) {
    const int k0 = kb * 32;

    /* issue ALL of this block's loads up front: K tiles first, then V tiles.
       QK^T only needs the K tiles, so the V loads stay outstanding through
       the whole softmax and are only waited on at the P*V WMMAs. */
    v16h bk[2][2];
#pragma unroll
    for (int sub = 0; sub < 2; ++sub)
#pragma unroll
      for (int s = 0; s < 2; ++s)
        bk[sub][s] = ld2g(Kb + (k0 + sub * 16 + l15) * DD + s * 32 + hi * 16, 8);
    v16h vb[4];
#pragma unroll
    for (int g = 0; g < 4; ++g)
      vb[g] = ld2g(Vb + (g * 16 + l15) * TT + k0 + hi * 8, 16);

    /* S = Q K^T : two 16x16 score tiles (keys k0..15, k0+16..31) */
    v8f sacc[2];
#pragma unroll
    for (int sub = 0; sub < 2; ++sub)
#pragma unroll
      for (int j = 0; j < 8; ++j) sacc[sub][j] = 0.0f;
#pragma unroll
    for (int sub = 0; sub < 2; ++sub)
#pragma unroll
      for (int s = 0; s < 2; ++s)
        sacc[sub] = wmma_f16(qa[s], bk[sub][s], sacc[sub]);

    /* scale + causal mask (wave-uniform predicate) */
    const bool need_mask = (k0 + 31) > q0;
#pragma unroll
    for (int sub = 0; sub < 2; ++sub) {
      const int key = k0 + sub * 16 + l15;
#pragma unroll
      for (int r = 0; r < 8; ++r) {
        const float sv  = sacc[sub][r] * 0.125f;   /* 1/sqrt(64) */
        const int   row = q0 + r + hi * 8;
        sacc[sub][r] = (need_mask && key > row) ? NEG_INF : sv;
      }
    }

    /* online softmax per row-slot; rows live across 16-lane halves */
#pragma unroll
    for (int r = 0; r < 8; ++r) {
      const float s0 = sacc[0][r], s1 = sacc[1][r];
      float mx = fmaxf(s0, s1);
      mx = fmaxf(mx, __shfl_xor(mx, 1, 32));
      mx = fmaxf(mx, __shfl_xor(mx, 2, 32));
      mx = fmaxf(mx, __shfl_xor(mx, 4, 32));
      mx = fmaxf(mx, __shfl_xor(mx, 8, 32));
      const float mnew = fmaxf(m8[r], mx);
      const bool  dead = (mnew == NEG_INF);
      const float alpha = dead ? 1.0f : __expf(m8[r] - mnew);
      const float p0 = dead ? 0.0f : __expf(s0 - mnew);
      const float p1 = dead ? 0.0f : __expf(s1 - mnew);
      float rs = p0 + p1;
      rs += __shfl_xor(rs, 1, 32);
      rs += __shfl_xor(rs, 2, 32);
      rs += __shfl_xor(rs, 4, 32);
      rs += __shfl_xor(rs, 8, 32);
      l8[r] = l8[r] * alpha + rs;
      m8[r] = mnew;
#pragma unroll
      for (int g = 0; g < 4; ++g) oacc[g][r] *= alpha;
      /* stage P (C/D layout -> memory row-major) into LDS */
      myp[(r + hi * 8) * 32 + l15]      = (half_t)p0;
      myp[(r + hi * 8) * 32 + 16 + l15] = (half_t)p1;
    }
    asm volatile("s_wait_dscnt 0" ::: "memory");

    /* reload P in A-matrix layout (LDS ops are in-order per wave) */
    v16h pa;
    {
      union { struct { v8h lo; v8h hi; } s; v16h v; } u;
      const half_t* pr = myp + l15 * 32 + hi * 8;
      u.s.lo = *(const v8h*)(pr);
      u.s.hi = *(const v8h*)(pr + 16);
      pa = u.v;
    }

    /* O += P V : V tiles were loaded at block start */
#pragma unroll
    for (int g = 0; g < 4; ++g)
      oacc[g] = wmma_f16(pa, vb[g], oacc[g]);
  }

  /* normalize + write context (f16) */
  const int b = bh / HH, h = bh % HH;
#pragma unroll
  for (int r = 0; r < 8; ++r) {
    const float inv = 1.0f / l8[r];
    const int t = q0 + r + hi * 8;
    half_t* crow = Ctx + (b * TT + t) * CC + h * DD;
#pragma unroll
    for (int g = 0; g < 4; ++g)
      crow[g * 16 + l15] = (half_t)(oacc[g][r] * inv);
  }
}

/* ---------------------------- output projection -------------------------- */
__global__ void __launch_bounds__(256) k_gemm_proj(const half_t* __restrict__ ws,
                                                   float* __restrict__ out) {
  const half_t* Ctx = ws + OFF_CTX;
  const half_t* Wph = ws + OFF_WPH;

  const int wid = blockIdx.x * (blockDim.x >> 5) + (threadIdx.x >> 5);
  const int tm = wid & 127;      /* 128 m-tiles (32 rows each) */
  const int tn = wid >> 7;       /* 12  n-tiles */
  const int m0 = tm * 32, n0 = tn * 64;

  v8f acc[8];
#pragma unroll
  for (int i = 0; i < 8; ++i)
#pragma unroll
    for (int j = 0; j < 8; ++j) acc[i][j] = 0.0f;

  gemm32x64<CC>(Ctx + m0 * CC, Wph + n0 * CC, acc);

  const int lane = threadIdx.x & 31, l15 = lane & 15, hi = lane >> 4;
#pragma unroll
  for (int mi = 0; mi < 2; ++mi)
#pragma unroll
    for (int g = 0; g < 4; ++g)
#pragma unroll
      for (int r = 0; r < 8; ++r)
        out[(m0 + mi * 16 + r + hi * 8) * CC + n0 + g * 16 + l15] =
            acc[mi * 4 + g][r];
}

extern "C" void kernel_launch(void* const* d_in, const int* in_sizes, int n_in,
                              void* d_out, int out_size, void* d_ws, size_t ws_size,
                              hipStream_t stream) {
  const float* x  = (const float*)d_in[0];
  const float* wa = (const float*)d_in[1];
  const float* wp = (const float*)d_in[2];
  half_t* ws  = (half_t*)d_ws;
  float*  out = (float*)d_out;

  k_convert<<<1024, 256, 0, stream>>>(x, wa, wp, ws);
  k_gemm_qkv<<<576, 256, 0, stream>>>(ws);       /* 4608 waves: 128x36 tiles */
  k_attention<<<384, 256, 0, stream>>>(ws);      /* 3072 waves: 24 heads x 128 q-tiles */
  k_gemm_proj<<<192, 256, 0, stream>>>(ws, out); /* 1536 waves: 128x12 tiles */
}
